// EarthSpecificBlock_26233660244005
// MI455X (gfx1250) — compile-verified, matrix operand order checked by reading, not run on hardware
//
#include <hip/hip_runtime.h>
#include <hip/hip_bf16.h>
#include <math.h>

// ---------------------------------------------------------------------------
// EarthSpecificBlock (Pangu/Swin 3D window attention) for MI455X / gfx1250.
// GEMMs: V_WMMA_F32_16X16X32_F16, B panels staged to LDS via TDM
// (tensor_load_to_lds + s_wait_tensorcnt), A fragments from global.
// ---------------------------------------------------------------------------

typedef __attribute__((ext_vector_type(16))) _Float16 v16h;
typedef __attribute__((ext_vector_type(8)))  _Float16 v8h;
typedef __attribute__((ext_vector_type(8)))  float    v8f;
typedef __attribute__((ext_vector_type(4)))  unsigned int v4u32;
typedef __attribute__((ext_vector_type(8)))  int      v8i32;
typedef __attribute__((ext_vector_type(4)))  int      v4i32;

// ---- problem constants ----
#define PL   8
#define LAT  96
#define LON  180
#define C    384
#define H    12
#define HD   32
#define NTOK (PL*LAT*LON)        // 138240
#define NWIN 960                 // 15 * 64
#define NTW  144                 // tokens per window
#define HID  1536
#define MT   (NTOK/16)           // 8640 M-tiles
#define MB8  (MT/8)              // 1080 M-blocks (8 tiles / block)

#if defined(__HIP_DEVICE_COMPILE__) && defined(__has_builtin)
#if __has_builtin(__builtin_amdgcn_tensor_load_to_lds) && __has_builtin(__builtin_amdgcn_s_wait_tensorcnt)
#define USE_TDM 1
#endif
#endif
#ifndef USE_TDM
#define USE_TDM 0
#endif

__device__ __forceinline__ v8f zero8() {
  v8f z;
#pragma unroll
  for (int i = 0; i < 8; ++i) z[i] = 0.0f;
  return z;
}

// 16-bit A-fragment (16x32), row-major source with `stride` halves.
// lanes 0-15: row = lane,    K = {0..7, 16..23}
// lanes 16-31: row = lane-16, K = {8..15, 24..31}   (ISA 05_wmma §7.12.2)
__device__ __forceinline__ v16h load_a_frag(const _Float16* base, int stride, int lane) {
  const int r = lane & 15, hi = lane >> 4;
  const _Float16* p = base + (size_t)r * stride + hi * 8;
  v16h a;
  v8h* ah = reinterpret_cast<v8h*>(&a);
  ah[0] = *reinterpret_cast<const v8h*>(p);
  ah[1] = *reinterpret_cast<const v8h*>(p + 16);
  return a;
}

// B-fragment from packed tile (512 contiguous halves, lane*16 each).
__device__ __forceinline__ v16h load_b_packed(const _Float16* tile, int lane) {
  const _Float16* p = tile + lane * 16;
  v16h b;
  v8h* bh = reinterpret_cast<v8h*>(&b);
  bh[0] = *reinterpret_cast<const v8h*>(p);
  bh[1] = *reinterpret_cast<const v8h*>(p + 8);
  return b;
}

#define WMMA_F16(a, b, c) \
  __builtin_amdgcn_wmma_f32_16x16x32_f16(false, (a), false, (b), (short)0, (c), false, false)

#if USE_TDM
// Tensor Data Mover: 2D tile load into LDS. Descriptor per 08_async_tensor §8:
// group0: [0]=count(1), [1]=lds_addr, [2]=gaddr.lo, [3]=gaddr.hi | type(2)<<30
// group1: data_size=3 (8B units), tensor_dim0=stride, tensor_dim1=rows,
//         tile_dim0 (16b), tile_dim1, tensor_dim0_stride.
// This toolchain exposes the 6-arg builtin: (v4u, v8i, v4i, v4i, v8i, cpol).
__device__ __forceinline__ void tdm_load_2d(void* ldsDst, const void* gSrc,
                                            unsigned tileRowU64, unsigned tileRows,
                                            unsigned long long strideU64) {
  unsigned long long ga = (unsigned long long)(uintptr_t)gSrc;
  unsigned lds = (unsigned)(uintptr_t)ldsDst;
  v4u32 g0 = { 1u, lds, (unsigned)ga,
               (unsigned)((ga >> 32) & 0x01ffffffu) | (2u << 30) };
  unsigned td0 = (unsigned)strideU64;      // tensor row length (8B units)
  unsigned td1 = tileRows;
  v8i32 g1;
  g1[0] = (int)(3u << 16);                               // data_size = 8 bytes
  g1[1] = (int)((td0 & 0xffffu) << 16);                  // tensor_dim0 lo
  g1[2] = (int)((td0 >> 16) | ((td1 & 0xffffu) << 16));  // dim0 hi | dim1 lo
  g1[3] = (int)((td1 >> 16) | (tileRowU64 << 16));       // dim1 hi | tile_dim0
  g1[4] = (int)(tileRows & 0xffffu);                     // tile_dim1 (tile_dim2=0)
  g1[5] = (int)(unsigned)(strideU64 & 0xffffffffull);    // dim0 stride lo
  g1[6] = (int)(unsigned)(strideU64 >> 32);              // dim0 stride hi
  g1[7] = 0;
  v4i32 z4 = { 0, 0, 0, 0 };
  v8i32 z8;
#pragma unroll
  for (int i = 0; i < 8; ++i) z8[i] = 0;
  __builtin_amdgcn_tensor_load_to_lds(g0, g1, z4, z4, z8, 0);
}
#endif

// roll(-1,-3,-6) + window_partition: windowed token id -> source token id
__device__ __forceinline__ int gather_src(int gwt) {
  int wi = gwt / NTW, n = gwt - wi * NTW;
  int wb = wi / 64,  w = wi - wb * 64;
  int npl = w >> 4,  nlat = w & 15;
  int wpl = n / 72,  rem = n - wpl * 72;
  int wlat = rem / 12, wlon = rem - wlat * 12;
  int pl = npl * 2 + wpl, la = nlat * 6 + wlat, lo = wb * 12 + wlon;
  int sp = (pl + 1) & 7;
  int sl = la + 3;  if (sl >= LAT) sl -= LAT;
  int so = lo + 6;  if (so >= LON) so -= LON;
  return (sp * LAT + sl) * LON + so;
}

// ---------------------------------------------------------------------------
// Weight packing: f32 [K x Nout] -> f16 WMMA B tiles (tile=(nt*Kt+kt)*512).
// ---------------------------------------------------------------------------
__global__ void __launch_bounds__(256)
k_pack_b(const float* __restrict__ w, _Float16* __restrict__ dst, int Kt, int Nout) {
  int gid = blockIdx.x * blockDim.x + threadIdx.x;
  int tile = gid >> 5, lane = gid & 31;
  int nt = tile / Kt, kt = tile - nt * Kt;
  int r = lane & 15, hi = lane >> 4;
  int n = nt * 16 + r;
  int kbase = kt * 32 + hi * 8;
  _Float16* o = dst + (size_t)tile * 512 + lane * 16;
#pragma unroll
  for (int j = 0; j < 16; ++j) {
    int k = kbase + ((j < 8) ? j : (j + 8));
    o[j] = (_Float16)w[(size_t)k * Nout + n];
  }
}

// ---------------------------------------------------------------------------
// LayerNorm (wave-per-token). GATHER fuses roll + window_partition.
// ---------------------------------------------------------------------------
template <bool GATHER>
__global__ void __launch_bounds__(256)
k_ln(const float* __restrict__ xin, const float* __restrict__ g,
     const float* __restrict__ be, _Float16* __restrict__ out) {
  int wv = threadIdx.x >> 5, lane = threadIdx.x & 31;
  int row = blockIdx.x * 8 + wv;
  int src = GATHER ? gather_src(row) : row;
  const float* xp = xin + (size_t)src * C;
  float v[12], s = 0.f, ss = 0.f;
#pragma unroll
  for (int j = 0; j < 12; ++j) {
    float t = xp[lane + j * 32];
    v[j] = t; s += t; ss += t * t;
  }
#pragma unroll
  for (int o = 16; o >= 1; o >>= 1) { s += __shfl_xor(s, o, 32); ss += __shfl_xor(ss, o, 32); }
  float mean = s * (1.0f / C);
  float inv  = rsqrtf(ss * (1.0f / C) - mean * mean + 1e-5f);
  _Float16* op = out + (size_t)row * C;
#pragma unroll
  for (int j = 0; j < 12; ++j) {
    int c = lane + j * 32;
    op[c] = (_Float16)((v[j] - mean) * inv * g[c] + be[c]);
  }
}

// ---------------------------------------------------------------------------
// Block GEMM core. Block = 8 waves; wave wv owns rows (mb*8+wv)*16..+15,
// all waves share a 64-column B panel staged in LDS (48KB chunk = 4 Ntiles x
// 12 Ktiles). KCHUNKS chunks of K (12 k-tiles each) with barriers between.
// A: one-ahead prefetch from global; B: ds_load_b128 from LDS.
// ---------------------------------------------------------------------------
template <int KCHUNKS>
__device__ __forceinline__ void gemm_block(const _Float16* __restrict__ A /* + row0*lda */,
                                           int lda,
                                           const _Float16* __restrict__ Bpanel,
                                           _Float16* __restrict__ Bsh,
                                           v8f acc[4]) {
  const int tid = threadIdx.x, lane = tid & 31;
  constexpr int KT = KCHUNKS * 12;
#pragma unroll
  for (int t = 0; t < 4; ++t) acc[t] = zero8();

  for (int kc = 0; kc < KCHUNKS; ++kc) {
    __syncthreads();                      // protect previous chunk reads
#if USE_TDM
    if (tid < 32) {                       // wave 0 issues the DMA
      tdm_load_2d(Bsh, Bpanel + (size_t)kc * 12 * 512,
                  /*tileRowU64=*/12 * 128, /*tileRows=*/4,
                  /*strideU64=*/(unsigned long long)KT * 128);
      __builtin_amdgcn_s_wait_tensorcnt(0);
    }
#else
    for (int idx = tid; idx < 4 * 768; idx += 256) {   // 768 v8h per N-tile row
      int rowt = idx / 768, inner = idx - rowt * 768;
      reinterpret_cast<v8h*>(Bsh)[idx] =
          *reinterpret_cast<const v8h*>(Bpanel + (size_t)rowt * KT * 512 +
                                        (size_t)kc * 12 * 512 + (size_t)inner * 8);
    }
#endif
    __syncthreads();

    v16h a = load_a_frag(A + (kc * 12) * 32, lda, lane);
#pragma unroll
    for (int kt = 0; kt < 12; ++kt) {
      v16h an = a;
      if (kt < 11) an = load_a_frag(A + (kc * 12 + kt + 1) * 32, lda, lane);
#pragma unroll
      for (int t = 0; t < 4; ++t) {
        v16h b = load_b_packed(Bsh + (t * 12 + kt) * 512, lane);
        acc[t] = WMMA_F16(a, b, acc[t]);
      }
      a = an;
    }
  }
}

// ---------------------------------------------------------------------------
// QKV GEMM: [NTOK x 384] @ [384 x 1152]; scatter into per-(window,head)
// [144 x 32] q/k/v buffers, q pre-scaled by HD^-0.5.
// ---------------------------------------------------------------------------
__global__ void __launch_bounds__(256)
k_qkv(const _Float16* __restrict__ xw, const _Float16* __restrict__ pw,
      const float* __restrict__ bq,
      _Float16* __restrict__ qb, _Float16* __restrict__ kb, _Float16* __restrict__ vb) {
  __shared__ __align__(16) _Float16 Bsh[4 * 12 * 512];
  int bid = blockIdx.x;
  int nb = bid / MB8, mb = bid - nb * MB8;
  int wv = threadIdx.x >> 5, lane = threadIdx.x & 31;
  int row0 = (mb * 8 + wv) * 16, n0 = nb * 64;
  v8f acc[4];
  gemm_block<1>(xw + (size_t)row0 * C, C, pw + (size_t)(n0 >> 4) * 12 * 512, Bsh, acc);
  int r = lane & 15, hi = lane >> 4;
#pragma unroll
  for (int t = 0; t < 4; ++t) {
    int j = n0 + t * 16 + r;
    int seg = j / C, jj = j - seg * C;
    int head = jj >> 5, d = jj & 31;
    float bias = bq[j];
    _Float16* dst = (seg == 0) ? qb : (seg == 1) ? kb : vb;
    float scl = (seg == 0) ? 0.17677669529663687f : 1.0f;   // 32^-0.5
#pragma unroll
    for (int i = 0; i < 8; ++i) {
      int row = row0 + i + hi * 8;
      int wi = row / NTW, n = row - wi * NTW;
      dst[((size_t)(wi * H + head) * NTW + n) * HD + d] = (_Float16)((acc[t][i] + bias) * scl);
    }
  }
}

// ---------------------------------------------------------------------------
// Attention: one block per (window, head). 9 waves; wave wv owns rows
// 16*wv..16*wv+15. S=QK^T (one K=32 WMMA/tile), +pos_bias +mask, softmax
// in registers, P staged to LDS (K padded 144->160), O=P@V via WMMA.
// ---------------------------------------------------------------------------
__global__ void __launch_bounds__(288)
k_attn(const _Float16* __restrict__ qb, const _Float16* __restrict__ kb,
       const _Float16* __restrict__ vb, const float* __restrict__ pos_bias,
       const float* __restrict__ amask, _Float16* __restrict__ aout) {
  __shared__ __align__(16) _Float16 Pls[NTW * 160];   // 46080 B
  __shared__ __align__(16) _Float16 Vt[HD * 160];     // 10240 B

  int task = blockIdx.x;
  int wi = task / H, h = task - wi * H;
  int wb = wi / 64, wloc = wi - wb * 64;
  int tid = threadIdx.x, wv = tid >> 5, lane = tid & 31;
  int r = lane & 15, hi = lane >> 4;

  const _Float16* Q = qb + (size_t)(wi * H + h) * NTW * HD;
  const _Float16* K = kb + (size_t)(wi * H + h) * NTW * HD;
  const _Float16* V = vb + (size_t)(wi * H + h) * NTW * HD;

  for (int idx = tid; idx < HD * 160; idx += 288) {
    int d = idx / 160, m = idx - d * 160;
    Vt[idx] = (m < NTW) ? V[(size_t)m * HD + d] : (_Float16)0.0f;
  }
  for (int idx = tid; idx < NTW * 16; idx += 288) {
    int row = idx >> 4, c = idx & 15;
    Pls[row * 160 + 144 + c] = (_Float16)0.0f;
  }

  v16h aQ = load_a_frag(Q + (size_t)(wv * 16) * HD, HD, lane);
  v8f s[9];
#pragma unroll
  for (int j = 0; j < 9; ++j) {
    v16h bK = load_a_frag(K + (size_t)(j * 16) * HD, HD, lane);
    s[j] = WMMA_F16(aQ, bK, zero8());
  }

  const float* pb = pos_bias + (size_t)(h * 64 + wloc) * NTW * NTW;
  const float* am = amask    + (size_t)(wb * 64 + wloc) * NTW * NTW;
#pragma unroll
  for (int j = 0; j < 9; ++j) {
    int m = j * 16 + r;
#pragma unroll
    for (int i = 0; i < 8; ++i) {
      int n = wv * 16 + i + hi * 8;
      s[j][i] += pb[(size_t)n * NTW + m] + am[(size_t)n * NTW + m];
    }
  }

#pragma unroll
  for (int i = 0; i < 8; ++i) {
    float mx = -1e30f;
#pragma unroll
    for (int j = 0; j < 9; ++j) mx = fmaxf(mx, s[j][i]);
#pragma unroll
    for (int o = 8; o >= 1; o >>= 1) mx = fmaxf(mx, __shfl_xor(mx, o, 32));
    float sum = 0.f;
#pragma unroll
    for (int j = 0; j < 9; ++j) { float e = __expf(s[j][i] - mx); s[j][i] = e; sum += e; }
#pragma unroll
    for (int o = 8; o >= 1; o >>= 1) sum += __shfl_xor(sum, o, 32);
    float inv = 1.0f / sum;
#pragma unroll
    for (int j = 0; j < 9; ++j) s[j][i] *= inv;
  }

#pragma unroll
  for (int j = 0; j < 9; ++j) {
    int m = j * 16 + r;
#pragma unroll
    for (int i = 0; i < 8; ++i) {
      int n = wv * 16 + i + hi * 8;
      Pls[n * 160 + m] = (_Float16)s[j][i];
    }
  }
  __syncthreads();

  v8f o0 = zero8(), o1 = zero8();
#pragma unroll
  for (int kt = 0; kt < 5; ++kt) {
    v16h aP = load_a_frag(&Pls[(wv * 16) * 160 + kt * 32], 160, lane);
    v16h b0 = load_a_frag(&Vt[0  * 160 + kt * 32], 160, lane);
    v16h b1 = load_a_frag(&Vt[16 * 160 + kt * 32], 160, lane);
    o0 = WMMA_F16(aP, b0, o0);
    o1 = WMMA_F16(aP, b1, o1);
  }

#pragma unroll
  for (int ct = 0; ct < 2; ++ct) {
    const v8f& oc = ct ? o1 : o0;
    int d = ct * 16 + r;
#pragma unroll
    for (int i = 0; i < 8; ++i) {
      int n = wv * 16 + i + hi * 8;
      aout[((size_t)wi * NTW + n) * C + h * HD + d] = (_Float16)oc[i];
    }
  }
}

// ---------------------------------------------------------------------------
// Proj GEMM + window_reverse + un-roll scatter + residual.
// ---------------------------------------------------------------------------
__global__ void __launch_bounds__(256)
k_proj(const _Float16* __restrict__ aout, const _Float16* __restrict__ pw,
       const float* __restrict__ bp, const float* __restrict__ x,
       float* __restrict__ xnew) {
  __shared__ __align__(16) _Float16 Bsh[4 * 12 * 512];
  int bid = blockIdx.x;
  int nb = bid / MB8, mb = bid - nb * MB8;
  int wv = threadIdx.x >> 5, lane = threadIdx.x & 31;
  int row0 = (mb * 8 + wv) * 16, n0 = nb * 64;
  v8f acc[4];
  gemm_block<1>(aout + (size_t)row0 * C, C, pw + (size_t)(n0 >> 4) * 12 * 512, Bsh, acc);
  int r = lane & 15, hi = lane >> 4;
  int src[8];
#pragma unroll
  for (int i = 0; i < 8; ++i) src[i] = gather_src(row0 + i + hi * 8);
#pragma unroll
  for (int t = 0; t < 4; ++t) {
    int col = n0 + t * 16 + r;
    float bias = bp[col];
#pragma unroll
    for (int i = 0; i < 8; ++i) {
      size_t o = (size_t)src[i] * C + col;
      xnew[o] = x[o] + acc[t][i] + bias;
    }
  }
}

// ---------------------------------------------------------------------------
// MLP1: +bias, exact GELU, f16 out.
// ---------------------------------------------------------------------------
__global__ void __launch_bounds__(256)
k_mlp1(const _Float16* __restrict__ h2, const _Float16* __restrict__ pw,
       const float* __restrict__ b1, _Float16* __restrict__ m1) {
  __shared__ __align__(16) _Float16 Bsh[4 * 12 * 512];
  int bid = blockIdx.x;
  int nb = bid / MB8, mb = bid - nb * MB8;
  int wv = threadIdx.x >> 5, lane = threadIdx.x & 31;
  int row0 = (mb * 8 + wv) * 16, n0 = nb * 64;
  v8f acc[4];
  gemm_block<1>(h2 + (size_t)row0 * C, C, pw + (size_t)(n0 >> 4) * 12 * 512, Bsh, acc);
  int r = lane & 15, hi = lane >> 4;
#pragma unroll
  for (int t = 0; t < 4; ++t) {
    int col = n0 + t * 16 + r;
    float bias = b1[col];
#pragma unroll
    for (int i = 0; i < 8; ++i) {
      int row = row0 + i + hi * 8;
      float v = acc[t][i] + bias;
      v = 0.5f * v * (1.0f + erff(v * 0.70710678118654752f));
      m1[(size_t)row * HID + col] = (_Float16)v;
    }
  }
}

// ---------------------------------------------------------------------------
// MLP2: K=1536 in 4 LDS chunks; + bias + xnew -> d_out (f32).
// ---------------------------------------------------------------------------
__global__ void __launch_bounds__(256)
k_mlp2(const _Float16* __restrict__ m1, const _Float16* __restrict__ pw,
       const float* __restrict__ b2, const float* __restrict__ xnew,
       float* __restrict__ out) {
  __shared__ __align__(16) _Float16 Bsh[4 * 12 * 512];
  int bid = blockIdx.x;
  int nb = bid / MB8, mb = bid - nb * MB8;
  int wv = threadIdx.x >> 5, lane = threadIdx.x & 31;
  int row0 = (mb * 8 + wv) * 16, n0 = nb * 64;
  v8f acc[4];
  gemm_block<4>(m1 + (size_t)row0 * HID, HID, pw + (size_t)(n0 >> 4) * 48 * 512, Bsh, acc);
  int r = lane & 15, hi = lane >> 4;
#pragma unroll
  for (int t = 0; t < 4; ++t) {
    int col = n0 + t * 16 + r;
    float bias = b2[col];
#pragma unroll
    for (int i = 0; i < 8; ++i) {
      int row = row0 + i + hi * 8;
      size_t o = (size_t)row * C + col;
      out[o] = xnew[o] + acc[t][i] + bias;
    }
  }
}

// ---------------------------------------------------------------------------
// host launch
// ---------------------------------------------------------------------------
extern "C" void kernel_launch(void* const* d_in, const int* in_sizes, int n_in,
                              void* d_out, int out_size, void* d_ws, size_t ws_size,
                              hipStream_t stream) {
  const float* x        = (const float*)d_in[0];
  const float* w_qkv    = (const float*)d_in[1];
  const float* b_qkv    = (const float*)d_in[2];
  const float* w_proj   = (const float*)d_in[3];
  const float* b_proj   = (const float*)d_in[4];
  const float* pos_bias = (const float*)d_in[5];
  const float* g1       = (const float*)d_in[6];
  const float* be1      = (const float*)d_in[7];
  const float* g2       = (const float*)d_in[8];
  const float* be2      = (const float*)d_in[9];
  const float* w1       = (const float*)d_in[10];
  const float* b1       = (const float*)d_in[11];
  const float* w2       = (const float*)d_in[12];
  const float* b2       = (const float*)d_in[13];
  const float* amask    = (const float*)d_in[14];
  float* out = (float*)d_out;

  constexpr size_t SZ_TOKH = (size_t)NTOK * C * 2;
  constexpr size_t SZ_TOKF = (size_t)NTOK * C * 4;
  constexpr size_t SZ_M1   = (size_t)NTOK * HID * 2;

  char* ws = (char*)d_ws;
  size_t off = 0;
  _Float16* xw    = (_Float16*)(ws + off); off += SZ_TOKH;
  _Float16* qb    = (_Float16*)(ws + off); off += SZ_TOKH;
  _Float16* kb    = (_Float16*)(ws + off); off += SZ_TOKH;
  _Float16* vbuf  = (_Float16*)(ws + off); off += SZ_TOKH;
  _Float16* aout  = (_Float16*)(ws + off); off += SZ_TOKH;
  float*    xnew  = (float*)   (ws + off); off += SZ_TOKF;
  _Float16* h2    = (_Float16*)(ws + off); off += SZ_TOKH;
  _Float16* m1    = (_Float16*)(ws + off); off += SZ_M1;
  _Float16* pwqkv = (_Float16*)(ws + off); off += (size_t)72 * 12 * 512 * 2;
  _Float16* pwprj = (_Float16*)(ws + off); off += (size_t)24 * 12 * 512 * 2;
  _Float16* pw1   = (_Float16*)(ws + off); off += (size_t)96 * 12 * 512 * 2;
  _Float16* pw2   = (_Float16*)(ws + off); off += (size_t)24 * 48 * 512 * 2;

  k_pack_b<<<108, 256, 0, stream>>>(w_qkv,  pwqkv, 12, 1152);
  k_pack_b<<< 36, 256, 0, stream>>>(w_proj, pwprj, 12, 384);
  k_pack_b<<<144, 256, 0, stream>>>(w1,     pw1,   12, 1536);
  k_pack_b<<<144, 256, 0, stream>>>(w2,     pw2,   48, 384);

  k_ln<true><<<NTOK / 8, 256, 0, stream>>>(x, g1, be1, xw);

  k_qkv<<<MB8 * 18, 256, 0, stream>>>(xw, pwqkv, b_qkv, qb, kb, vbuf);

  k_attn<<<NWIN * H, 288, 0, stream>>>(qb, kb, vbuf, pos_bias, amask, aout);

  k_proj<<<MB8 * 6, 256, 0, stream>>>(aout, pwprj, b_proj, x, xnew);

  k_ln<false><<<NTOK / 8, 256, 0, stream>>>(xnew, g2, be2, h2);

  k_mlp1<<<MB8 * 24, 256, 0, stream>>>(h2, pw1, b1, m1);
  k_mlp2<<<MB8 * 6, 256, 0, stream>>>(m1, pw2, b2, xnew, out);
}
